// EGT_Layer_68178310857270
// MI455X (gfx1250) — compile-verified, matrix operand order checked by reading, not run on hardware
//
#include <hip/hip_runtime.h>
#include <hip/hip_bf16.h>
#include <math.h>

// ---------------------------------------------------------------------------
// EGT layer for MI455X (gfx1250): wave32, bf16 WMMA 16x16x32, fp32 accumulate.
// ---------------------------------------------------------------------------

typedef __bf16 bf16;
typedef __attribute__((ext_vector_type(16))) __bf16 v16bf;
typedef __attribute__((ext_vector_type(8)))  float  v8f;

#define B_   8
#define N_   256
#define D_   768
#define EW_  64
#define H_   32
#define DD_  24
#define SCALE_ 0.20412414523193154f   // 24^-0.5
#define NTOK (B_ * N_)                // 2048
#define HEADPLANE (N_ * N_)           // 65536
#define QSEC ((size_t)B_ * H_ * N_ * DD_)   // elems per q/k/v head-major section

// ---------------- helpers ----------------

__device__ __forceinline__ bf16 f2bf(float x) {
  union { float f; unsigned u; } a; a.f = x;
  unsigned r = a.u + 0x7FFFu + ((a.u >> 16) & 1u);
  union { unsigned short s; bf16 b; } c; c.s = (unsigned short)(r >> 16);
  return c.b;
}
__device__ __forceinline__ float bf2f(bf16 x) {
  union { unsigned short s; bf16 b; } c; c.b = x;
  union { float f; unsigned u; } a; a.u = ((unsigned)c.s) << 16;
  return a.f;
}
// hardware v_rcp_f32 (avoid precise-division expansion)
__device__ __forceinline__ float fast_rcp(float x) { return __builtin_amdgcn_rcpf(x); }
__device__ __forceinline__ float sigmoid_f(float x) {
  return fast_rcp(1.0f + __expf(-x));
}
// tanh-approx gelu via the identity 0.5*(1+tanh(y)) == sigmoid(2y)
__device__ __forceinline__ float gelu_f(float x) {
  float y = 1.5957691216f * (x + 0.044715f * x * x * x);  // 2*0.7978845608
  return x * fast_rcp(1.0f + __expf(-y));
}
__device__ __forceinline__ v8f v8f_zero() {
  v8f z;
#pragma unroll
  for (int i = 0; i < 8; ++i) z[i] = 0.0f;
  return z;
}

// A/B fragment from LDS staged as [row-or-col][K], stride elems (16B aligned).
// Per-lane: r = lane&15 ; K set = {khi..khi+7, 16+khi..16+khi+7}, khi = 8*(lane>=16).
// Two ds_load_b128 per fragment.
__device__ __forceinline__ v16bf load_frag(const bf16* base, int stride) {
  int lane = threadIdx.x & 31;
  int r    = lane & 15;
  int khi  = (lane >> 4) << 3;
  const bf16* p = base + r * stride;
  union { v16bf v; uint4 q[2]; } u;
  u.q[0] = *(const uint4*)(p + khi);
  u.q[1] = *(const uint4*)(p + 16 + khi);
  return u.v;
}

// Fragment from fp32 LDS (convert while packing): layout [row][K], stride elems.
__device__ __forceinline__ v16bf load_frag_f32(const float* base, int stride) {
  int lane = threadIdx.x & 31;
  int r    = lane & 15;
  int khi  = (lane >> 4) << 3;
  const float* p = base + r * stride;
  v16bf f;
#pragma unroll
  for (int e = 0; e < 8; ++e) f[e] = f2bf(p[khi + e]);
#pragma unroll
  for (int e = 0; e < 8; ++e) f[8 + e] = f2bf(p[16 + khi + e]);
  return f;
}

// B fragment when LDS is stored [K][n] (transposed relative to load_frag).
__device__ __forceinline__ v16bf load_frag_T(const bf16* base, int stride) {
  int lane = threadIdx.x & 31;
  int n    = lane & 15;
  int khi  = (lane >> 4) << 3;
  v16bf f;
#pragma unroll
  for (int e = 0; e < 8; ++e) f[e] = base[(khi + e) * stride + n];
#pragma unroll
  for (int e = 0; e < 8; ++e) f[8 + e] = base[(16 + khi + e) * stride + n];
  return f;
}

#define WMMA_BF16(A, Bm, C) \
  __builtin_amdgcn_wmma_f32_16x16x32_bf16(false, (A), false, (Bm), (short)0, (C), false, false)

// ---------------- generic kernels ----------------

__global__ __launch_bounds__(256) void f32_to_bf16_k(const float* __restrict__ in,
                                                     bf16* __restrict__ out, int n) {
  int i = blockIdx.x * 256 + threadIdx.x;
  if (i < n) out[i] = f2bf(in[i]);
}

// LayerNorm rows of 768, fp32 in -> bf16 out. One block per row.
__global__ __launch_bounds__(256) void ln768_to_bf16(const float* __restrict__ x,
                                                     const float* __restrict__ g,
                                                     const float* __restrict__ bta,
                                                     bf16* __restrict__ out) {
  __shared__ float red[256];
  int row = blockIdx.x, t = threadIdx.x;
  const float* xr = x + (size_t)row * D_;
  float v[3], s = 0.0f;
#pragma unroll
  for (int i = 0; i < 3; ++i) { v[i] = xr[t + i * 256]; s += v[i]; }
  red[t] = s; __syncthreads();
  for (int off = 128; off > 0; off >>= 1) { if (t < off) red[t] += red[t + off]; __syncthreads(); }
  float mean = red[0] * (1.0f / D_);
  __syncthreads();
  float s2 = 0.0f;
#pragma unroll
  for (int i = 0; i < 3; ++i) { float d = v[i] - mean; s2 += d * d; }
  red[t] = s2; __syncthreads();
  for (int off = 128; off > 0; off >>= 1) { if (t < off) red[t] += red[t + off]; __syncthreads(); }
  float rstd = rsqrtf(red[0] * (1.0f / D_) + 1e-5f);
#pragma unroll
  for (int i = 0; i < 3; ++i) {
    int c = t + i * 256;
    out[(size_t)row * D_ + c] = f2bf((v[i] - mean) * rstd * g[c] + bta[c]);
  }
}

// Generic 128x128-tile bf16 WMMA GEMM, fp32 accumulate.
// EPI 0: qkv -> head-major scatter outB[sec][(b*32+h)*256+l][d], Q scaled
// EPI 1: fp32 out = acc + bias + res (residual)
// EPI 2: bf16 out = gelu(acc + bias)
template <int EPI>
__global__ __launch_bounds__(256) void gemm_bf16_128x128(
    const bf16* __restrict__ A, const bf16* __restrict__ Bw,
    const float* __restrict__ bias, const float* __restrict__ res,
    float* __restrict__ outF, bf16* __restrict__ outB,
    int M, int N, int K, float scale) {
  __shared__ bf16 As[128 * 40];
  __shared__ bf16 Bt[128 * 40];
  int t = threadIdx.x, w = t >> 5, lane = t & 31;
  int wm = w >> 1, wn = w & 1;
  int n0 = blockIdx.x * 128, m0 = blockIdx.y * 128;
  v8f acc[2][4];
#pragma unroll
  for (int i = 0; i < 2; ++i)
#pragma unroll
    for (int j = 0; j < 4; ++j) acc[i][j] = v8f_zero();

  int ar = t >> 1, ak = (t & 1) * 16;
  for (int kc = 0; kc < K; kc += 32) {
    // A tile: two b128 loads per thread
    const uint4* ap = (const uint4*)(A + (size_t)(m0 + ar) * K + kc + ak);
    *(uint4*)&As[ar * 40 + ak]     = ap[0];
    *(uint4*)&As[ar * 40 + ak + 8] = ap[1];
    // B tile: vectorized b128 loads, transpose on LDS-write side
#pragma unroll
    for (int i = 0; i < 2; ++i) {
      int id = i * 256 + t;       // 512 jobs: 32 k x 16 n-groups-of-8
      int k  = id >> 4;
      int ng = id & 15;
      union { uint4 q; bf16 b[8]; } u;
      u.q = *(const uint4*)(Bw + (size_t)(kc + k) * N + n0 + ng * 8);
#pragma unroll
      for (int j = 0; j < 8; ++j) Bt[(ng * 8 + j) * 40 + k] = u.b[j];
    }
    if (kc + 32 < K) {  // global_prefetch_b8 for the next K chunk
      __builtin_prefetch(A + (size_t)(m0 + ar) * K + kc + 32 + ak, 0, 1);
      __builtin_prefetch(Bw + (size_t)(kc + 32 + (t >> 3)) * N + n0 + (t & 7) * 16, 0, 1);
    }
    __syncthreads();
#pragma unroll
    for (int rt = 0; rt < 2; ++rt) {
      v16bf a = load_frag(&As[(wm * 32 + rt * 16) * 40], 40);
#pragma unroll
      for (int ct = 0; ct < 4; ++ct) {
        v16bf bb = load_frag(&Bt[(wn * 64 + ct * 16) * 40], 40);
        acc[rt][ct] = WMMA_BF16(a, bb, acc[rt][ct]);
      }
    }
    __syncthreads();
  }
  int col = lane & 15, rb = (lane >> 4) << 3;
#pragma unroll
  for (int rt = 0; rt < 2; ++rt)
#pragma unroll
    for (int ct = 0; ct < 4; ++ct)
#pragma unroll
      for (int r = 0; r < 8; ++r) {
        int row = m0 + wm * 32 + rt * 16 + rb + r;
        int c   = n0 + wn * 64 + ct * 16 + col;
        float v = acc[rt][ct][r] + bias[c];
        if (EPI == 0) {
          int sec = c / D_;
          int cc  = c - sec * D_;
          int d = cc >> 5, hh = cc & 31;
          int bb2 = row >> 8, l = row & 255;
          if (sec == 0) v *= scale;
          outB[(size_t)sec * QSEC +
               ((size_t)((bb2 * H_ + hh) * N_ + l)) * DD_ + d] = f2bf(v);
        } else if (EPI == 1) {
          outF[(size_t)row * N + c] = v + res[(size_t)row * N + c];
        } else {
          outB[(size_t)row * N + c] = f2bf(gelu_f(v));
        }
      }
}

// ---------------- edge: fused LN(e) + eg GEMM -> E, gates (head-major) -------

__global__ __launch_bounds__(256) void edge_ln_eg(
    const float* __restrict__ e, const float* __restrict__ mask,
    const float* __restrict__ lng, const float* __restrict__ lnb,
    const bf16* __restrict__ Weg, const float* __restrict__ beg,
    bf16* __restrict__ Eg, bf16* __restrict__ Gg) {
  __shared__ bf16 Elds[128 * 72];
  __shared__ bf16 Wt[64 * 72];
  int t = threadIdx.x, w = t >> 5, lane = t & 31;
  int m0 = blockIdx.x * 128, l = blockIdx.y, b = blockIdx.z;
#pragma unroll
  for (int i = 0; i < 16; ++i) {
    int id = i * 256 + t;
    int n = id & 63, k = id >> 6;
    Wt[n * 72 + k] = Weg[k * 64 + n];
  }
  if (t < 128) {
    const float* er = e + ((size_t)(b * N_ + l) * N_ + m0 + t) * EW_;
    float s = 0.0f, s2 = 0.0f;
#pragma unroll
    for (int c = 0; c < EW_; ++c) { float v = er[c]; s += v; s2 += v * v; }
    float mean = s * (1.0f / EW_);
    float var  = s2 * (1.0f / EW_) - mean * mean;
    float rstd = rsqrtf(var + 1e-5f);
#pragma unroll
    for (int c = 0; c < EW_; ++c)
      Elds[t * 72 + c] = f2bf((er[c] - mean) * rstd * lng[c] + lnb[c]);
  }
  __syncthreads();
  int wm = w >> 1, wn = w & 1;
  v8f acc[2][2];
#pragma unroll
  for (int i = 0; i < 2; ++i)
#pragma unroll
    for (int j = 0; j < 2; ++j) acc[i][j] = v8f_zero();
#pragma unroll
  for (int kcc = 0; kcc < 64; kcc += 32)
#pragma unroll
    for (int rt = 0; rt < 2; ++rt) {
      v16bf a = load_frag(&Elds[(wm * 32 + rt * 16) * 72 + kcc], 72);
#pragma unroll
      for (int ct = 0; ct < 2; ++ct) {
        v16bf bb = load_frag(&Wt[(wn * 32 + ct * 16) * 72 + kcc], 72);
        acc[rt][ct] = WMMA_BF16(a, bb, acc[rt][ct]);
      }
    }
  int col = lane & 15, rb = (lane >> 4) << 3;
#pragma unroll
  for (int rt = 0; rt < 2; ++rt)
#pragma unroll
    for (int ct = 0; ct < 2; ++ct)
#pragma unroll
      for (int r = 0; r < 8; ++r) {
        int rl = wm * 32 + rt * 16 + rb + r;
        int c  = wn * 32 + ct * 16 + col;
        int m  = m0 + rl;
        float val = acc[rt][ct][r] + beg[c];
        if (c < H_) {
          Eg[(size_t)(b * H_ + c) * HEADPLANE + l * N_ + m] = f2bf(val);
        } else {
          int hh = c - H_;
          float mv = mask[(size_t)b * HEADPLANE + l * N_ + m];
          Gg[(size_t)(b * H_ + hh) * HEADPLANE + l * N_ + m] = f2bf(sigmoid_f(val + mv));
        }
      }
}

// ---------------- attention: one workgroup per (b,h) ------------------------

__global__ __launch_bounds__(256) void egt_attention(
    const bf16* __restrict__ qhm, const bf16* __restrict__ Eg,
    const bf16* __restrict__ Gg, const float* __restrict__ mask,
    bf16* __restrict__ Hhat, bf16* __restrict__ vatt) {
  __shared__ bf16  Ks[N_ * 40];       // [m][k<=31] padded DD 24->32
  __shared__ bf16  Vs[N_ * 40];
  __shared__ bf16  Qs[128 * 40];
  __shared__ float Ssm[128 * 264];    // scores fp32, half the l-rows at a time
  __shared__ float degs[128];

  int t = threadIdx.x, w = t >> 5, lane = t & 31;
  int b = blockIdx.x >> 5, h = blockIdx.x & 31;
  size_t gbase = (size_t)(b * H_ + h) * HEADPLANE;
  size_t mbase = (size_t)b * HEADPLANE;
  size_t hrow  = (size_t)(b * H_ + h) * N_;
  const uint4 z4 = make_uint4(0u, 0u, 0u, 0u);

  // K,V for this head: contiguous DD=24 bf16 per row (3x b128), zero-pad 24..31
  {
    const uint4* kp = (const uint4*)(qhm + QSEC     + (hrow + t) * DD_);
    const uint4* vp = (const uint4*)(qhm + 2 * QSEC + (hrow + t) * DD_);
    *(uint4*)&Ks[t * 40 + 0]  = kp[0];
    *(uint4*)&Ks[t * 40 + 8]  = kp[1];
    *(uint4*)&Ks[t * 40 + 16] = kp[2];
    *(uint4*)&Ks[t * 40 + 24] = z4;
    *(uint4*)&Vs[t * 40 + 0]  = vp[0];
    *(uint4*)&Vs[t * 40 + 8]  = vp[1];
    *(uint4*)&Vs[t * 40 + 16] = vp[2];
    *(uint4*)&Vs[t * 40 + 24] = z4;
  }
  __syncthreads();

  for (int lh = 0; lh < 2; ++lh) {
    int l0 = lh * 128;
    if (t < 128) {  // Q (pre-scaled) for 128 rows
      const uint4* qp = (const uint4*)(qhm + (hrow + l0 + t) * DD_);
      *(uint4*)&Qs[t * 40 + 0]  = qp[0];
      *(uint4*)&Qs[t * 40 + 8]  = qp[1];
      *(uint4*)&Qs[t * 40 + 16] = qp[2];
      *(uint4*)&Qs[t * 40 + 24] = z4;
    }
    __syncthreads();

    // S = Q K^T + E  (single K=32 WMMA per 16x16 tile); store H_hat, keep S+mask in LDS
    {
      v16bf a = load_frag(&Qs[(w * 16) * 40], 40);
      int col = lane & 15, rb = (lane >> 4) << 3;
#pragma unroll
      for (int ct = 0; ct < 16; ++ct) {
        v16bf bb = load_frag(&Ks[(ct * 16) * 40], 40);
        v8f acc = WMMA_BF16(a, bb, v8f_zero());
#pragma unroll
        for (int r = 0; r < 8; ++r) {
          int lr = w * 16 + rb + r;
          int m  = ct * 16 + col;
          size_t gi = gbase + (size_t)(l0 + lr) * N_ + m;
          float v = acc[r] + bf2f(Eg[gi]);
          Hhat[gi] = f2bf(v);
          Ssm[lr * 264 + m] = v + mask[mbase + (size_t)(l0 + lr) * N_ + m];
        }
      }
    }
    __syncthreads();

    // softmax over m, multiply by gates, accumulate degree scalers
    for (int rr = 0; rr < 16; ++rr) {
      int r = w * 16 + rr;
      int l = l0 + r;
      float vals[8], mx = -3.4e38f;
#pragma unroll
      for (int j = 0; j < 8; ++j) {
        vals[j] = Ssm[r * 264 + j * 32 + lane];
        mx = fmaxf(mx, vals[j]);
      }
      for (int off = 16; off > 0; off >>= 1) mx = fmaxf(mx, __shfl_xor(mx, off, 32));
      float psum = 0.0f, gsum = 0.0f, pg[8];
#pragma unroll
      for (int j = 0; j < 8; ++j) {
        float p = __expf(vals[j] - mx);
        psum += p;
        float g = bf2f(Gg[gbase + (size_t)l * N_ + j * 32 + lane]);
        gsum += g;
        pg[j] = p * g;
      }
      for (int off = 16; off > 0; off >>= 1) {
        psum += __shfl_xor(psum, off, 32);
        gsum += __shfl_xor(gsum, off, 32);
      }
      float inv = fast_rcp(psum);
#pragma unroll
      for (int j = 0; j < 8; ++j) Ssm[r * 264 + j * 32 + lane] = pg[j] * inv;
      if (lane == 0) degs[r] = log1pf(gsum);
    }
    __syncthreads();

    // V_att = A @ V, scale by degree, scatter into (DD,H)-interleaved layout
    {
      int col = lane & 15, rb = (lane >> 4) << 3;
#pragma unroll
      for (int ct = 0; ct < 2; ++ct) {
        v8f acc = v8f_zero();
#pragma unroll
        for (int kc = 0; kc < 8; ++kc) {
          v16bf a  = load_frag_f32(&Ssm[(w * 16) * 264 + kc * 32], 264);
          v16bf bb = load_frag_T(&Vs[(kc * 32) * 40 + ct * 16], 40);
          acc = WMMA_BF16(a, bb, acc);
        }
#pragma unroll
        for (int r = 0; r < 8; ++r) {
          int lr   = w * 16 + rb + r;
          int kcol = ct * 16 + col;
          if (kcol < DD_) {
            float v = acc[r] * degs[lr];
            vatt[(size_t)(b * N_ + l0 + lr) * D_ + kcol * H_ + h] = f2bf(v);
          }
        }
      }
    }
    __syncthreads();
  }
}

// ---------------- edge tail: e_att + residual + LN + FFN, fully fused -------

__global__ __launch_bounds__(256) void edge_tail(
    const bf16* __restrict__ Hhat, const float* __restrict__ e,
    const bf16* __restrict__ Woe, const float* __restrict__ boe,
    const float* __restrict__ lng, const float* __restrict__ lnb,
    const bf16* __restrict__ eW1, const float* __restrict__ eb1,
    const bf16* __restrict__ eW2, const float* __restrict__ eb2,
    float* __restrict__ eout) {
  __shared__ bf16  Hh[128 * 40];      // [m][h] K=32
  __shared__ bf16  Wtoe[64 * 40];     // [n][k]
  __shared__ float e2s[128 * 64];     // residual e2 (fp32)
  __shared__ bf16  Eln[128 * 72];
  __shared__ bf16  Wt1[64 * 72];
  __shared__ bf16  Wt2[64 * 72];
  __shared__ bf16  T1[128 * 72];

  int t = threadIdx.x, w = t >> 5, lane = t & 31;
  int m0 = blockIdx.x * 128, l = blockIdx.y, b = blockIdx.z;
  int wm = w >> 1, wn = w & 1;
  int col = lane & 15, rb = (lane >> 4) << 3;

  // stage H_hat tile (transpose gather h-major -> [m][h]) and weights
#pragma unroll
  for (int i = 0; i < 16; ++i) {
    int id = i * 256 + t;
    int m = id & 127, hh = id >> 7;
    Hh[m * 40 + hh] = Hhat[(size_t)(b * H_ + hh) * HEADPLANE + l * N_ + m0 + m];
  }
#pragma unroll
  for (int i = 0; i < 8; ++i) {
    int id = i * 256 + t;
    int n = id & 63, k = id >> 6;
    Wtoe[n * 40 + k] = Woe[k * 64 + n];
  }
#pragma unroll
  for (int i = 0; i < 16; ++i) {
    int id = i * 256 + t;
    int n = id & 63, k = id >> 6;
    Wt1[n * 72 + k] = eW1[k * 64 + n];
    Wt2[n * 72 + k] = eW2[k * 64 + n];
  }
  __syncthreads();

  // e2 = e + H_hat @ W_oe + b_oe
#pragma unroll
  for (int rt = 0; rt < 2; ++rt) {
    v16bf a = load_frag(&Hh[(wm * 32 + rt * 16) * 40], 40);
#pragma unroll
    for (int ct = 0; ct < 2; ++ct) {
      v16bf bb = load_frag(&Wtoe[(wn * 32 + ct * 16) * 40], 40);
      v8f acc = WMMA_BF16(a, bb, v8f_zero());
#pragma unroll
      for (int r = 0; r < 8; ++r) {
        int rl = wm * 32 + rt * 16 + rb + r;
        int c  = wn * 32 + ct * 16 + col;
        float ein = e[((size_t)(b * N_ + l) * N_ + m0 + rl) * EW_ + c];
        e2s[rl * 64 + c] = ein + acc[r] + boe[c];
      }
    }
  }
  __syncthreads();

  // LN rows of e2
  if (t < 128) {
    float s = 0.0f, s2 = 0.0f;
#pragma unroll
    for (int c = 0; c < EW_; ++c) { float v = e2s[t * 64 + c]; s += v; s2 += v * v; }
    float mean = s * (1.0f / EW_);
    float rstd = rsqrtf(s2 * (1.0f / EW_) - mean * mean + 1e-5f);
#pragma unroll
    for (int c = 0; c < EW_; ++c)
      Eln[t * 72 + c] = f2bf((e2s[t * 64 + c] - mean) * rstd * lng[c] + lnb[c]);
  }
  __syncthreads();

  // T1 = gelu(Eln @ eW1 + eb1)
#pragma unroll
  for (int rt = 0; rt < 2; ++rt)
#pragma unroll
    for (int ct = 0; ct < 2; ++ct) {
      v8f acc = v8f_zero();
#pragma unroll
      for (int kcc = 0; kcc < 64; kcc += 32) {
        v16bf a  = load_frag(&Eln[(wm * 32 + rt * 16) * 72 + kcc], 72);
        v16bf bb = load_frag(&Wt1[(wn * 32 + ct * 16) * 72 + kcc], 72);
        acc = WMMA_BF16(a, bb, acc);
      }
#pragma unroll
      for (int r = 0; r < 8; ++r) {
        int rl = wm * 32 + rt * 16 + rb + r;
        int c  = wn * 32 + ct * 16 + col;
        T1[rl * 72 + c] = f2bf(gelu_f(acc[r] + eb1[c]));
      }
    }
  __syncthreads();

  // e_out = e2 + T1 @ eW2 + eb2
#pragma unroll
  for (int rt = 0; rt < 2; ++rt)
#pragma unroll
    for (int ct = 0; ct < 2; ++ct) {
      v8f acc = v8f_zero();
#pragma unroll
      for (int kcc = 0; kcc < 64; kcc += 32) {
        v16bf a  = load_frag(&T1[(wm * 32 + rt * 16) * 72 + kcc], 72);
        v16bf bb = load_frag(&Wt2[(wn * 32 + ct * 16) * 72 + kcc], 72);
        acc = WMMA_BF16(a, bb, acc);
      }
#pragma unroll
      for (int r = 0; r < 8; ++r) {
        int rl = wm * 32 + rt * 16 + rb + r;
        int c  = wn * 32 + ct * 16 + col;
        eout[((size_t)(b * N_ + l) * N_ + m0 + rl) * EW_ + c] =
            e2s[rl * 64 + c] + acc[r] + eb2[c];
      }
    }
}

// ---------------- host launch ----------------

extern "C" void kernel_launch(void* const* d_in, const int* in_sizes, int n_in,
                              void* d_out, int out_size, void* d_ws, size_t ws_size,
                              hipStream_t stream) {
  const float* h        = (const float*)d_in[0];
  const float* e        = (const float*)d_in[1];
  const float* mask     = (const float*)d_in[2];
  const float* ln_h_g   = (const float*)d_in[3];
  const float* ln_h_b   = (const float*)d_in[4];
  const float* ln_e_g   = (const float*)d_in[5];
  const float* ln_e_b   = (const float*)d_in[6];
  const float* W_qkv    = (const float*)d_in[7];
  const float* b_qkv    = (const float*)d_in[8];
  const float* W_eg     = (const float*)d_in[9];
  const float* b_eg     = (const float*)d_in[10];
  const float* W_oh     = (const float*)d_in[11];
  const float* b_oh     = (const float*)d_in[12];
  const float* W_oe     = (const float*)d_in[13];
  const float* b_oe     = (const float*)d_in[14];
  const float* nln_g    = (const float*)d_in[15];
  const float* nln_b    = (const float*)d_in[16];
  const float* nW1      = (const float*)d_in[17];
  const float* nb1      = (const float*)d_in[18];
  const float* nW2      = (const float*)d_in[19];
  const float* nb2      = (const float*)d_in[20];
  const float* eln_g    = (const float*)d_in[21];
  const float* eln_b    = (const float*)d_in[22];
  const float* eW1      = (const float*)d_in[23];
  const float* eb1      = (const float*)d_in[24];
  const float* eW2      = (const float*)d_in[25];
  const float* eb2      = (const float*)d_in[26];

  char* wsp = (char*)d_ws;
  auto alloc = [&](size_t bytes) -> char* {
    char* p = wsp;
    wsp += (bytes + 255) & ~(size_t)255;
    return p;
  };
  bf16* Wqkv_bf = (bf16*)alloc((size_t)D_ * 3 * D_ * 2);
  bf16* Weg_bf  = (bf16*)alloc((size_t)EW_ * 2 * H_ * 2);
  bf16* Woh_bf  = (bf16*)alloc((size_t)D_ * D_ * 2);
  bf16* Woe_bf  = (bf16*)alloc((size_t)H_ * EW_ * 2);
  bf16* nW1_bf  = (bf16*)alloc((size_t)D_ * D_ * 2);
  bf16* nW2_bf  = (bf16*)alloc((size_t)D_ * D_ * 2);
  bf16* eW1_bf  = (bf16*)alloc((size_t)EW_ * EW_ * 2);
  bf16* eW2_bf  = (bf16*)alloc((size_t)EW_ * EW_ * 2);
  bf16* hln_bf  = (bf16*)alloc((size_t)NTOK * D_ * 2);
  bf16* qhm_bf  = (bf16*)alloc(3 * QSEC * 2);           // head-major Q,K,V
  bf16* E_bf    = (bf16*)alloc((size_t)B_ * H_ * HEADPLANE * 2);
  bf16* G_bf    = (bf16*)alloc((size_t)B_ * H_ * HEADPLANE * 2);
  bf16* Hh_bf   = (bf16*)alloc((size_t)B_ * H_ * HEADPLANE * 2);
  bf16* vatt_bf = (bf16*)alloc((size_t)NTOK * D_ * 2);
  float* h2_f   = (float*)alloc((size_t)NTOK * D_ * 4);
  bf16* hln2_bf = (bf16*)alloc((size_t)NTOK * D_ * 2);
  bf16* t1_bf   = (bf16*)alloc((size_t)NTOK * D_ * 2);

  auto cvt = [&](const float* src, bf16* dst, int n) {
    f32_to_bf16_k<<<(n + 255) / 256, 256, 0, stream>>>(src, dst, n);
  };
  cvt(W_qkv, Wqkv_bf, D_ * 3 * D_);
  cvt(W_eg,  Weg_bf,  EW_ * 2 * H_);
  cvt(W_oh,  Woh_bf,  D_ * D_);
  cvt(W_oe,  Woe_bf,  H_ * EW_);
  cvt(nW1,   nW1_bf,  D_ * D_);
  cvt(nW2,   nW2_bf,  D_ * D_);
  cvt(eW1,   eW1_bf,  EW_ * EW_);
  cvt(eW2,   eW2_bf,  EW_ * EW_);

  // node pre-LN + qkv projection (Q pre-scaled, head-major layout)
  ln768_to_bf16<<<NTOK, 256, 0, stream>>>(h, ln_h_g, ln_h_b, hln_bf);
  gemm_bf16_128x128<0><<<dim3((3 * D_) / 128, NTOK / 128), 256, 0, stream>>>(
      hln_bf, Wqkv_bf, b_qkv, nullptr, nullptr, qhm_bf,
      NTOK, 3 * D_, D_, SCALE_);

  // edge LN + eg projection -> E, gates (head-major)
  edge_ln_eg<<<dim3(2, N_, B_), 256, 0, stream>>>(
      e, mask, ln_e_g, ln_e_b, Weg_bf, b_eg, E_bf, G_bf);

  // attention per (b,h)
  egt_attention<<<B_ * H_, 256, 0, stream>>>(qhm_bf, E_bf, G_bf, mask, Hh_bf, vatt_bf);

  // node output projection + residual, LN, FFN
  gemm_bf16_128x128<1><<<dim3(D_ / 128, NTOK / 128), 256, 0, stream>>>(
      vatt_bf, Woh_bf, b_oh, h, h2_f, nullptr, NTOK, D_, D_, 1.0f);
  ln768_to_bf16<<<NTOK, 256, 0, stream>>>(h2_f, nln_g, nln_b, hln2_bf);
  gemm_bf16_128x128<2><<<dim3(D_ / 128, NTOK / 128), 256, 0, stream>>>(
      hln2_bf, nW1_bf, nb1, nullptr, nullptr, t1_bf, NTOK, D_, D_, 1.0f);
  gemm_bf16_128x128<1><<<dim3(D_ / 128, NTOK / 128), 256, 0, stream>>>(
      t1_bf, nW2_bf, nb2, h2_f, (float*)d_out, nullptr, NTOK, D_, D_, 1.0f);

  // edge tail (e_att + residual + LN + FFN) -> d_out e-section
  float* e_out = (float*)d_out + (size_t)NTOK * D_;
  edge_tail<<<dim3(2, N_, B_), 256, 0, stream>>>(
      Hh_bf, e, Woe_bf, b_oe, eln_g, eln_b, eW1_bf, eb1, eW2_bf, eb2, e_out);
}
